// ExpertAttention_49177375539824
// MI455X (gfx1250) — matrix operand order repaired and checked by reading, not compiled, not run on hardware
//
#include <hip/hip_runtime.h>

// ---------------------------------------------------------------------------
// MoE LoRA attention for MI455X (gfx1250, wave32, WMMA bf16 f32-accum,
// TDM async staging of K/V into LDS).
// ---------------------------------------------------------------------------

typedef __attribute__((ext_vector_type(16))) __bf16 v16bf;
typedef __attribute__((ext_vector_type(8)))  __bf16 v8bf;
typedef __attribute__((ext_vector_type(8)))  float  v8f;
typedef __attribute__((ext_vector_type(4)))  unsigned int v4u;
typedef __attribute__((ext_vector_type(8)))  int  v8i;
typedef __attribute__((ext_vector_type(4)))  int  v4i;

#define DD   768
#define SS   512
#define BBB  32
#define HHH  12
#define DHH  64
#define RRR  128
#define NTOK (BBB * SS)            // 16384
#define WMAT ((size_t)DD * DD)     // 589824

__device__ __forceinline__ __bf16 f2bf(float f) {
  unsigned u = __builtin_bit_cast(unsigned, f);
  u += 0x7FFFu + ((u >> 16) & 1u);           // round-to-nearest-even
  unsigned short h = (unsigned short)(u >> 16);
  return __builtin_bit_cast(__bf16, h);
}

// ---------------------------------------------------------------------------
// Tensor Data Mover: 1-row 2D descriptor copying n8 * 8 bytes from global to
// LDS.  D# bitfields per CDNA5 ISA.
// Toolchain uses the 6-arg builtin (clang-23 / therock-10.0 headers).
// ---------------------------------------------------------------------------
__device__ __forceinline__ void tdm_copy_1d(unsigned lds_byte_addr,
                                            const void* gptr, unsigned n8) {
  unsigned long long ga = (unsigned long long)(uintptr_t)gptr;
  v4u g0;
  g0[0] = 1u;                                          // count=1, user mode
  g0[1] = lds_byte_addr;                               // lds_addr
  g0[2] = (unsigned)(ga & 0xFFFFFFFFu);                // global_addr[31:0]
  g0[3] = (unsigned)((ga >> 32) & 0x01FFFFFFu) | (2u << 30); // addr[56:32]|type=2
  v8i g1;
  g1[0] = (int)(3u << 16);                             // wg_mask=0, data_size=8B
  g1[1] = (int)((n8 & 0xFFFFu) << 16);                 // tensor_dim0 lo
  g1[2] = (int)((n8 >> 16) & 0xFFFFu) | (int)(1u << 16); // t_dim0 hi | t_dim1=1
  g1[3] = (int)((n8 & 0xFFFFu) << 16);                 // t_dim1 hi=0 | tile_dim0
  g1[4] = 1;                                           // tile_dim1=1, tile_dim2=0
  g1[5] = (int)n8;                                     // tensor_dim0_stride lo
  g1[6] = 0;                                           // stride hi | dim1_stride lo
  g1[7] = 0;
  v4i z4 = {0, 0, 0, 0};
  v8i z8 = {0, 0, 0, 0, 0, 0, 0, 0};
  __builtin_amdgcn_tensor_load_to_lds(g0, g1, z4, z4, z8, 0);
}

// ---------------------------------------------------------------------------
// One wave computes a 32x64 f32 tile:  A (32 x kdim bf16 row-major) times
// Bt (B transposed: Bt[n][k] bf16 row-major).  2 A-frags x 4 B-frags ->
// 8 WMMA per K=32 step (1.5 b128 loads per WMMA).  Layouts per ISA 7.12.2.
// ---------------------------------------------------------------------------
__device__ __forceinline__ void wave_gemm_32x64(const __bf16* __restrict__ A, int lda,
                                                const __bf16* __restrict__ Bt, int ldb,
                                                int kdim, v8f acc[2][4]) {
  const int lane = threadIdx.x & 31;
  const int half = lane >> 4;
  const int mr   = lane & 15;
  const __bf16* a0 = A + (size_t)mr * lda + half * 16;
  const __bf16* a1 = A + (size_t)(16 + mr) * lda + half * 16;
  const __bf16* b0 = Bt + (size_t)mr * ldb + half * 16;
#pragma unroll 2
  for (int kb = 0; kb < kdim; kb += 32) {
    union { v16bf v; v8bf h[2]; } ax[2];
    ax[0].h[0] = *(const v8bf*)(a0 + kb);
    ax[0].h[1] = *(const v8bf*)(a0 + kb + 8);
    ax[1].h[0] = *(const v8bf*)(a1 + kb);
    ax[1].h[1] = *(const v8bf*)(a1 + kb + 8);
    v16bf bv[4];
#pragma unroll
    for (int n = 0; n < 4; ++n)
      bv[n] = *(const v16bf*)(b0 + (size_t)(n * 16) * ldb + kb);
    __builtin_prefetch(b0 + kb + 256, 0, 1);
#pragma unroll
    for (int m = 0; m < 2; ++m)
#pragma unroll
      for (int n = 0; n < 4; ++n)
        acc[m][n] = __builtin_amdgcn_wmma_f32_16x16x32_bf16(
            false, ax[m].v, false, bv[n], (short)0, acc[m][n], false, false);
  }
}

// ---------------------------------------------------------------------------
// Kernel 1: routing.  mean over S -> encoder -> switch -> argmax.
// scale = rp_max/stop_grad(rp_max) == 1.0 exactly, so only routes matter.
// ---------------------------------------------------------------------------
__global__ void k_routing(const float* __restrict__ x,
                          const float* __restrict__ encW,
                          const float* __restrict__ encB,
                          const float* __restrict__ swW,
                          const float* __restrict__ swB,
                          int* __restrict__ routes, float* __restrict__ mcr) {
  __shared__ float mean[DD];
  __shared__ float henc[RRR];
  const int b = blockIdx.x, tid = threadIdx.x;
  for (int d = tid; d < DD; d += blockDim.x) {
    float s = 0.f;
    const float* xb = x + ((size_t)b * SS) * DD + d;
    for (int t = 0; t < SS; ++t) s += xb[(size_t)t * DD];
    mean[d] = s * (1.0f / SS);
  }
  __syncthreads();
  for (int j = tid; j < RRR; j += blockDim.x) {
    float s = encB[j];
    for (int d = 0; d < DD; ++d) s += mean[d] * encW[d * RRR + j];
    henc[j] = s;
  }
  __syncthreads();
  if (tid == 0) {
    float l0 = swB[0], l1 = swB[1];
    for (int r = 0; r < RRR; ++r) { l0 += henc[r] * swW[r * 2 + 0]; l1 += henc[r] * swW[r * 2 + 1]; }
    routes[b] = (l1 > l0) ? 1 : 0;   // jnp.argmax: first index wins ties
    if (b == 0) mcr[0] = 0.0f;
  }
}

// ---------------------------------------------------------------------------
// Kernel 2: f32 -> bf16 conversion of hidden_states.
// ---------------------------------------------------------------------------
__global__ void k_convert(const float* __restrict__ x, __bf16* __restrict__ xbf, int n4) {
  int i = blockIdx.x * blockDim.x + threadIdx.x;
  const int stride = gridDim.x * blockDim.x;
  for (; i < n4; i += stride) {
    float4 f = ((const float4*)x)[i];
    __bf16* o = xbf + (size_t)i * 4;
    o[0] = f2bf(f.x); o[1] = f2bf(f.y); o[2] = f2bf(f.z); o[3] = f2bf(f.w);
  }
}

// ---------------------------------------------------------------------------
// Kernel 3: build 12 transposed bf16 weight matrices (LoRA folded in).
//   wt[m][n][k] = W[k][n] + sum_r A[k][r]*Bm[r][n]
// ---------------------------------------------------------------------------
struct WPrepArgs {
  const float* W[12];
  const float* A[12];
  const float* Bm[12];
  __bf16* wt;
};

__global__ void k_wprep(WPrepArgs args) {
  const int n = blockIdx.y * 16 + threadIdx.x;   // output col (coalesced x)
  const int k = blockIdx.x * 16 + threadIdx.y;   // output row
  const int m = blockIdx.z;
  float v = args.W[m][(size_t)k * DD + n];
  if (args.A[m]) {
    const float* a  = args.A[m] + (size_t)k * RRR;
    const float* bm = args.Bm[m];
    float s = 0.f;
    for (int r = 0; r < RRR; ++r) s += a[r] * bm[(size_t)r * DD + n];
    v += s;
  }
  args.wt[(size_t)m * WMAT + (size_t)n * DD + k] = f2bf(v);
}

// ---------------------------------------------------------------------------
// Kernel 4: QKV projection.  grid(m_tiles=512, h=12, which=3), 1 wave/block,
// each wave does a 32x64 tile.  Since DD/64 == HHH, blockIdx.y == head.
//   Q: [B,H,S,dh] bf16, scaled by 1/sqrt(dh); K: [B,H,S,dh]; V: [B,H,dh,S].
// ---------------------------------------------------------------------------
struct QkvArgs {
  const __bf16* xbf; const __bf16* wt; const int* routes;
  __bf16 *Q, *K, *Vt;
  const float* bc[3];      // common bq,bk,bv
  const float* be[2][3];   // expert bq,bk,bv
  int variant;             // 0 = common, 1 = routed expert
};

__global__ void k_qkv(QkvArgs args) {
  const int mt = blockIdx.x, h = blockIdx.y, which = blockIdx.z;
  const int tok0 = mt * 32;
  const int b = tok0 >> 9;
  const int e = args.routes[b];
  const int mat = (args.variant == 0) ? which : (4 + 4 * e + which);
  const float* bias = (args.variant == 0) ? args.bc[which] : args.be[e][which];

  const __bf16* A  = args.xbf + (size_t)tok0 * DD;
  const __bf16* Bt = args.wt + (size_t)mat * WMAT + (size_t)(h * 64) * DD;
  v8f acc[2][4];
#pragma unroll
  for (int m = 0; m < 2; ++m)
#pragma unroll
    for (int n = 0; n < 4; ++n) acc[m][n] = (v8f){0.f,0.f,0.f,0.f,0.f,0.f,0.f,0.f};
  wave_gemm_32x64(A, DD, Bt, DD, DD, acc);

  const int lane = threadIdx.x & 31, half = lane >> 4, mr = lane & 15;
  const size_t bh = (size_t)(b * HHH + h);
#pragma unroll
  for (int m = 0; m < 2; ++m)
#pragma unroll
    for (int n4 = 0; n4 < 4; ++n4) {
      const int dh = n4 * 16 + mr;           // 0..63 within this head
      const float bv = bias[h * 64 + dh];
#pragma unroll
      for (int i = 0; i < 8; ++i) {
        const int s = (tok0 + m * 16 + i + half * 8) & (SS - 1);
        const float val = acc[m][n4][i] + bv;
        if (which == 0)      args.Q[(bh * SS + s) * DHH + dh] = f2bf(val * 0.125f);
        else if (which == 1) args.K[(bh * SS + s) * DHH + dh] = f2bf(val);
        else                 args.Vt[(bh * DHH + dh) * SS + s] = f2bf(val);
      }
    }
}

// ---------------------------------------------------------------------------
// Kernel 5: fused attention per (b,h).  K,V^T staged into LDS by the Tensor
// Data Mover (each of 4 waves moves a 16KB chunk of each), two-pass softmax,
// probs@V via WMMA.  Never materializes the SxS score tensor in global.
// ---------------------------------------------------------------------------
__global__ void k_fa(const __bf16* __restrict__ Q, const __bf16* __restrict__ K,
                     const __bf16* __restrict__ Vt, __bf16* __restrict__ ctx,
                     const int* __restrict__ amask) {
  extern __shared__ char smem[];
  __bf16* Ks    = (__bf16*)smem;                               // [512][64]
  __bf16* Vs    = Ks + SS * DHH;                               // [64][512]
  float*  maskS = (float*)(smem + 2 * SS * DHH * 2);           // [512]
  __bf16* probs = (__bf16*)(smem + 2 * SS * DHH * 2 + SS * 4); // [4][16][512]

  const int bh = blockIdx.x;
  const int b = bh / HHH, h = bh - b * HHH;
  const int tid = threadIdx.x;
  const int lane = tid & 31, wave = tid >> 5;
  const int half = lane >> 4, mr = lane & 15;

  const __bf16* Kg = K  + (size_t)bh * SS * DHH;
  const __bf16* Vg = Vt + (size_t)bh * DHH * SS;

  // --- TDM async staging: each wave moves 16KB of K and 16KB of V^T ---
  {
    const unsigned ldsK = (unsigned)(uintptr_t)(void*)Ks;
    const unsigned ldsV = (unsigned)(uintptr_t)(void*)Vs;
    const unsigned chunkB = SS * DHH * 2 / 4;   // 16384 bytes per wave
    tdm_copy_1d(ldsK + wave * chunkB, (const char*)Kg + wave * chunkB, chunkB / 8);
    tdm_copy_1d(ldsV + wave * chunkB, (const char*)Vg + wave * chunkB, chunkB / 8);
  }
  for (int i = tid; i < SS; i += blockDim.x)
    maskS[i] = (1.0f - (float)amask[b * SS + i]) * -10000.0f;
  __builtin_amdgcn_s_wait_tensorcnt(0);
  __syncthreads();

  __bf16* Pw = probs + (size_t)wave * 16 * SS;

  for (int qt = wave; qt < SS / 16; qt += 4) {
    // ---- Q tile, A-operand layout (dh=64 -> two K=32 slabs) ----
    const __bf16* Qg = Q + ((size_t)bh * SS + qt * 16) * DHH;
    union { v16bf v; v8bf h2[2]; } qa[2];
#pragma unroll
    for (int kt = 0; kt < 2; ++kt) {
      const __bf16* p = Qg + (size_t)mr * DHH + kt * 32 + half * 16;
      qa[kt].h2[0] = *(const v8bf*)p;
      qa[kt].h2[1] = *(const v8bf*)(p + 8);
    }
    // ---- pass 1: row max of (QK^T + mask) ----
    float rmax[8];
#pragma unroll
    for (int i = 0; i < 8; ++i) rmax[i] = -3.0e38f;
    for (int nt = 0; nt < SS / 16; ++nt) {
      v8f sc = {0.f,0.f,0.f,0.f,0.f,0.f,0.f,0.f};
#pragma unroll
      for (int kt = 0; kt < 2; ++kt) {
        v16bf bt = *(const v16bf*)(Ks + (size_t)(nt * 16 + mr) * DHH + kt * 32 + half * 16);
        sc = __builtin_amdgcn_wmma_f32_16x16x32_bf16(false, qa[kt].v, false, bt,
                                                     (short)0, sc, false, false);
      }
      const float mv = maskS[nt * 16 + mr];
#pragma unroll
      for (int i = 0; i < 8; ++i) rmax[i] = fmaxf(rmax[i], sc[i] + mv);
    }
#pragma unroll
    for (int m = 1; m < 16; m <<= 1)
#pragma unroll
      for (int i = 0; i < 8; ++i) rmax[i] = fmaxf(rmax[i], __shfl_xor(rmax[i], m, 32));

    // ---- pass 2: exp, row sums, bf16 probs into LDS (row-major) ----
    float rsum[8];
#pragma unroll
    for (int i = 0; i < 8; ++i) rsum[i] = 0.f;
    for (int nt = 0; nt < SS / 16; ++nt) {
      v8f sc = {0.f,0.f,0.f,0.f,0.f,0.f,0.f,0.f};
#pragma unroll
      for (int kt = 0; kt < 2; ++kt) {
        v16bf bt = *(const v16bf*)(Ks + (size_t)(nt * 16 + mr) * DHH + kt * 32 + half * 16);
        sc = __builtin_amdgcn_wmma_f32_16x16x32_bf16(false, qa[kt].v, false, bt,
                                                     (short)0, sc, false, false);
      }
      const float mv = maskS[nt * 16 + mr];
#pragma unroll
      for (int i = 0; i < 8; ++i) {
        float e = __expf(sc[i] + mv - rmax[i]);
        rsum[i] += e;
        Pw[(size_t)(i + half * 8) * SS + nt * 16 + mr] = f2bf(e);
      }
    }
#pragma unroll
    for (int m = 1; m < 16; m <<= 1)
#pragma unroll
      for (int i = 0; i < 8; ++i) rsum[i] += __shfl_xor(rsum[i], m, 32);

    // ---- pass 3: ctx = P @ V, normalize, store bf16 [tok][D] ----
    v8f cacc[4];
#pragma unroll
    for (int n4 = 0; n4 < 4; ++n4) cacc[n4] = (v8f){0.f,0.f,0.f,0.f,0.f,0.f,0.f,0.f};
    for (int kb = 0; kb < SS; kb += 32) {
      union { v16bf v; v8bf h2[2]; } pa;
      pa.h2[0] = *(const v8bf*)(Pw + (size_t)mr * SS + kb + half * 16);
      pa.h2[1] = *(const v8bf*)(Pw + (size_t)mr * SS + kb + half * 16 + 8);
#pragma unroll
      for (int n4 = 0; n4 < 4; ++n4) {
        v16bf bt = *(const v16bf*)(Vs + (size_t)(n4 * 16 + mr) * SS + kb + half * 16);
        cacc[n4] = __builtin_amdgcn_wmma_f32_16x16x32_bf16(false, pa.v, false, bt,
                                                           (short)0, cacc[n4], false, false);
      }
    }
    float inv[8];
#pragma unroll
    for (int i = 0; i < 8; ++i) inv[i] = 1.0f / rsum[i];
#pragma unroll
    for (int n4 = 0; n4 < 4; ++n4)
#pragma unroll
      for (int i = 0; i < 8; ++i) {
        const int r = i + half * 8;
        const size_t tok = (size_t)b * SS + qt * 16 + r;
        ctx[tok * DD + h * DHH + n4 * 16 + mr] = f2bf(cacc[n4][i] * inv[i]);
      }
  }
}

// ---------------------------------------------------------------------------
// Kernel 6: output projection (+bias); variant 0 writes, variant 1 adds.
// grid(m_tiles=512, n_tiles=12), 1 wave/block, 32x64 tile per wave.
// ---------------------------------------------------------------------------
struct OutArgs {
  const __bf16* ctx; const __bf16* wt; const int* routes;
  float* out;
  const float* bo_c; const float* bo_e[2];
  int variant;
};

__global__ void k_out(OutArgs args) {
  const int mt = blockIdx.x, nt = blockIdx.y;
  const int tok0 = mt * 32;
  const int b = tok0 >> 9;
  const int e = args.routes[b];
  const int mat = (args.variant == 0) ? 3 : (4 + 4 * e + 3);
  const float* bias = (args.variant == 0) ? args.bo_c : args.bo_e[e];

  const __bf16* A  = args.ctx + (size_t)tok0 * DD;
  const __bf16* Bt = args.wt + (size_t)mat * WMAT + (size_t)(nt * 64) * DD;
  v8f acc[2][4];
#pragma unroll
  for (int m = 0; m < 2; ++m)
#pragma unroll
    for (int n = 0; n < 4; ++n) acc[m][n] = (v8f){0.f,0.f,0.f,0.f,0.f,0.f,0.f,0.f};
  wave_gemm_32x64(A, DD, Bt, DD, DD, acc);

  const int lane = threadIdx.x & 31, half = lane >> 4, mr = lane & 15;
#pragma unroll
  for (int m = 0; m < 2; ++m)
#pragma unroll
    for (int n4 = 0; n4 < 4; ++n4) {
      const int col = nt * 64 + n4 * 16 + mr;
      const float bv = bias[col];
#pragma unroll
      for (int i = 0; i < 8; ++i) {
        const size_t tok = (size_t)tok0 + m * 16 + i + half * 8;
        const float val = acc[m][n4][i] + bv;
        if (args.variant == 0) args.out[tok * DD + col] = val;
        else                   args.out[tok * DD + col] += val;
      }
    }
}

// ---------------------------------------------------------------------------
// Host launcher.  Input flattening (setup_inputs insertion order):
//   0 hidden_states, 1 attention_mask,
//   experts[e] (e=0,1): Wq,bq,Wk,bk,Wv,bv,Wo,bo,Aq,Bq,Av,Bv  -> 2+12e .. 13+12e
//   common: Wq 26, bq 27, Wk 28, bk 29, Wv 30, bv 31, Wo 32, bo 33
//   enc_W 34, enc_b 35, sw_W 36, sw_b 37
// ---------------------------------------------------------------------------
extern "C" void kernel_launch(void* const* d_in, const int* in_sizes, int n_in,
                              void* d_out, int out_size, void* d_ws, size_t ws_size,
                              hipStream_t stream) {
  (void)in_sizes; (void)n_in; (void)out_size; (void)ws_size;
  const float* x    = (const float*)d_in[0];
  const int*   mask = (const int*)d_in[1];
  auto F = [&](int i) { return (const float*)d_in[i]; };

  char* ws = (char*)d_ws;
  size_t off = 0;
  int* routes = (int*)ws;                         off = 256;
  __bf16* xbf = (__bf16*)(ws + off);              off += (size_t)NTOK * DD * 2;
  __bf16* wt  = (__bf16*)(ws + off);              off += (size_t)12 * WMAT * 2;
  __bf16* Q   = (__bf16*)(ws + off);              off += (size_t)NTOK * DD * 2;
  __bf16* K   = (__bf16*)(ws + off);              off += (size_t)NTOK * DD * 2;
  __bf16* Vt  = (__bf16*)(ws + off);              off += (size_t)NTOK * DD * 2;
  __bf16* ctx = (__bf16*)(ws + off);              off += (size_t)NTOK * DD * 2;

  float* out = (float*)d_out;
  float* mcr = out + (size_t)NTOK * DD;

  k_routing<<<BBB, 256, 0, stream>>>(x, F(34), F(35), F(36), F(37), routes, mcr);
  k_convert<<<4096, 256, 0, stream>>>(x, xbf, NTOK * DD / 4);

  // weight prep: 0-3 common q,k,v,o; 4-7 expert0; 8-11 expert1 (LoRA on q,v)
  static const int wIdx[12] = {26, 28, 30, 32,  2, 4, 6, 8,  14, 16, 18, 20};
  static const int aIdx[12] = {-1, -1, -1, -1, 10, -1, 12, -1, 22, -1, 24, -1};
  static const int bIdx[12] = {-1, -1, -1, -1, 11, -1, 13, -1, 23, -1, 25, -1};
  WPrepArgs wa;
  for (int m = 0; m < 12; ++m) {
    wa.W[m]  = F(wIdx[m]);
    wa.A[m]  = (aIdx[m] >= 0) ? F(aIdx[m]) : nullptr;
    wa.Bm[m] = (bIdx[m] >= 0) ? F(bIdx[m]) : nullptr;
  }
  wa.wt = wt;
  k_wprep<<<dim3(48, 48, 12), dim3(16, 16), 0, stream>>>(wa);

  const size_t faSmem = (size_t)2 * SS * DHH * 2 + SS * 4 + (size_t)4 * 16 * SS * 2;

  for (int variant = 0; variant < 2; ++variant) {
    QkvArgs qa;
    qa.xbf = xbf; qa.wt = wt; qa.routes = routes;
    qa.Q = Q; qa.K = K; qa.Vt = Vt;
    qa.bc[0] = F(27); qa.bc[1] = F(29); qa.bc[2] = F(31);
    qa.be[0][0] = F(3);  qa.be[0][1] = F(5);  qa.be[0][2] = F(7);
    qa.be[1][0] = F(15); qa.be[1][1] = F(17); qa.be[1][2] = F(19);
    qa.variant = variant;
    k_qkv<<<dim3(NTOK / 32, HHH, 3), 32, 0, stream>>>(qa);

    k_fa<<<BBB * HHH, 128, faSmem, stream>>>(Q, K, Vt, ctx, mask);

    OutArgs oa;
    oa.ctx = ctx; oa.wt = wt; oa.routes = routes; oa.out = out;
    oa.bo_c = F(33); oa.bo_e[0] = F(9); oa.bo_e[1] = F(21);
    oa.variant = variant;
    k_out<<<dim3(NTOK / 32, DD / 64), 32, 0, stream>>>(oa);
  }
}
